// SpikeNet_LSTM_55456617726551
// MI455X (gfx1250) — compile-verified, matrix operand order checked by reading, not run on hardware
//
#include <hip/hip_runtime.h>

// ---------------------------------------------------------------------------
// SLSTM (2-layer spiking LSTM) for MI455X / gfx1250.
// Persistent kernel, wave32 WMMA bf16 GEMMs, device-wide barrier.
//   B=256, T=512, D=256, H=512, O=128
// Weights staged ONCE into LDS in exact WMMA-fragment order and served from
// the 320KB LDS for all 512 timesteps. 32 blocks x 512 threads (16 waves):
// block = h-tile, wave = b-tile; each wave computes all 4 gate tiles
// (i,f,g,o) so the LSTM pointwise update is wave-local. A/B fragments are
// software-pipelined (next-k loads issued before current-k WMMAs).
// ---------------------------------------------------------------------------

constexpr int Bn = 256;
constexpr int Tn = 512;
constexpr int Dn = 256;
constexpr int Hn = 512;
constexpr int On = 128;
constexpr int NWG = 32;     // persistent workgroups (must be co-resident)
constexpr int NTH = 512;    // 16 waves / WG -> 512 waves total

// LDS layout (bytes): phase-A weight fragments, then phase-B weight fragments
constexpr int KS1 = (Dn + Hn) / 32;            // 24 k-steps, layer 1
constexpr int KS2 = (Hn + On) / 32;            // 20 k-steps, layer 2
constexpr int LDSA_BYTES = KS1 * 4 * 32 * 32;  // 98304
constexpr int LDSB_BYTES = KS2 * 4 * 32 * 32;  // 81920
constexpr int LDS_BYTES = LDSA_BYTES + LDSB_BYTES;  // 180224 (< 320KB)

typedef __attribute__((ext_vector_type(16))) __bf16 v16bf;
typedef __attribute__((ext_vector_type(8)))  __bf16 v8bf;
typedef __attribute__((ext_vector_type(8)))  float  v8f;
typedef __attribute__((ext_vector_type(4)))  float  v4f;

__device__ __forceinline__ float sigm(float x) { return 1.0f / (1.0f + __expf(-x)); }

// Device-wide generation barrier (persistent-kernel global sync).
__device__ __forceinline__ void grid_barrier(int* cnt, int* gen, int target) {
  __syncthreads();
  if (threadIdx.x == 0) {
    __threadfence();
    int prev = __hip_atomic_fetch_add(cnt, 1, __ATOMIC_ACQ_REL, __HIP_MEMORY_SCOPE_AGENT);
    if (prev == NWG - 1) {
      __hip_atomic_store(cnt, 0, __ATOMIC_RELAXED, __HIP_MEMORY_SCOPE_AGENT);
      __hip_atomic_fetch_add(gen, 1, __ATOMIC_RELEASE, __HIP_MEMORY_SCOPE_AGENT);
    } else {
      while (__hip_atomic_load(gen, __ATOMIC_ACQUIRE, __HIP_MEMORY_SCOPE_AGENT) < target) {
        __builtin_amdgcn_s_sleep(1);
      }
    }
    __threadfence();
  }
  __syncthreads();
}

// Build a 16-element bf16 fragment from two contiguous 8-elem (16B) chunks.
// Per CDNA5 16-bit A/B layouts: lane L (hl = L>=16) covers
//   chunk0: k0 + hl*8 + [0..7],   chunk1: k0 + 16 + hl*8 + [0..7]
__device__ __forceinline__ v16bf frag_from_bf16(const __bf16* row, int k0, int hl) {
  v8bf lo = *(const v8bf*)(row + k0 + hl * 8);
  v8bf hi = *(const v8bf*)(row + k0 + 16 + hl * 8);
  return __builtin_shufflevector(lo, hi, 0,1,2,3,4,5,6,7,8,9,10,11,12,13,14,15);
}

__device__ __forceinline__ v16bf frag_from_f32(const float* row, int k0, int hl) {
  const float* p = row + k0 + hl * 8;
  v4f x0 = *(const v4f*)(p);
  v4f x1 = *(const v4f*)(p + 4);
  v4f x2 = *(const v4f*)(p + 16);
  v4f x3 = *(const v4f*)(p + 20);
  v16bf a;
#pragma unroll
  for (int e = 0; e < 4; ++e) {
    a[e]      = (__bf16)x0[e];
    a[4 + e]  = (__bf16)x1[e];
    a[8 + e]  = (__bf16)x2[e];
    a[12 + e] = (__bf16)x3[e];
  }
  return a;
}

__global__ void __launch_bounds__(256) slstm_init(
    const float* __restrict__ Wih1, const float* __restrict__ Whh1,
    const float* __restrict__ bih1, const float* __restrict__ bhh1,
    const float* __restrict__ Wih2, const float* __restrict__ Whh2,
    const float* __restrict__ bih2, const float* __restrict__ bhh2,
    __bf16* __restrict__ Wih1b, __bf16* __restrict__ Whh1b,
    __bf16* __restrict__ Wih2b, __bf16* __restrict__ Whh2b,
    float* __restrict__ bias1, float* __restrict__ bias2,
    float* __restrict__ syn1, float* __restrict__ mem1f, __bf16* __restrict__ mem1b0,
    float* __restrict__ syn2, float* __restrict__ mem2f, __bf16* __restrict__ mem2b0,
    int* __restrict__ bar) {
  const size_t n0 = (size_t)4 * Hn * Dn;
  const size_t n1 = (size_t)4 * Hn * Hn;
  const size_t n2 = (size_t)4 * On * Hn;
  const size_t n3 = (size_t)4 * On * On;
  const size_t nb1 = 4 * Hn, nb2 = 4 * On;
  const size_t s1 = (size_t)Bn * Hn, s2 = (size_t)Bn * On;
  const size_t TOTAL = n0 + n1 + n2 + n3 + nb1 + nb2 + 3 * s1 + 3 * s2 + 2;
  size_t i = (size_t)blockIdx.x * blockDim.x + threadIdx.x;
  const size_t stride = (size_t)gridDim.x * blockDim.x;
  for (; i < TOTAL; i += stride) {
    size_t k = i;
    if (k < n0) { Wih1b[k] = (__bf16)Wih1[k]; continue; } k -= n0;
    if (k < n1) { Whh1b[k] = (__bf16)Whh1[k]; continue; } k -= n1;
    if (k < n2) { Wih2b[k] = (__bf16)Wih2[k]; continue; } k -= n2;
    if (k < n3) { Whh2b[k] = (__bf16)Whh2[k]; continue; } k -= n3;
    if (k < nb1) { bias1[k] = bih1[k] + bhh1[k]; continue; } k -= nb1;
    if (k < nb2) { bias2[k] = bih2[k] + bhh2[k]; continue; } k -= nb2;
    if (k < s1) { syn1[k] = 0.0f; continue; } k -= s1;
    if (k < s1) { mem1f[k] = 0.0f; continue; } k -= s1;
    if (k < s1) { mem1b0[k] = (__bf16)0.0f; continue; } k -= s1;
    if (k < s2) { syn2[k] = 0.0f; continue; } k -= s2;
    if (k < s2) { mem2f[k] = 0.0f; continue; } k -= s2;
    if (k < s2) { mem2b0[k] = (__bf16)0.0f; continue; } k -= s2;
    if (k < 2) { bar[k] = 0; }
  }
}

__global__ void __launch_bounds__(NTH) slstm_persistent(
    const float* __restrict__ x,
    const __bf16* __restrict__ Wih1b, const __bf16* __restrict__ Whh1b,
    const __bf16* __restrict__ Wih2b, const __bf16* __restrict__ Whh2b,
    const float* __restrict__ bias1, const float* __restrict__ bias2,
    float* __restrict__ syn1, float* __restrict__ mem1f,
    __bf16* __restrict__ mem1b0, __bf16* __restrict__ mem1b1,
    __bf16* __restrict__ spk1b,
    float* __restrict__ syn2, float* __restrict__ mem2f,
    __bf16* __restrict__ mem2b0, __bf16* __restrict__ mem2b1,
    const float* __restrict__ th1p, const float* __restrict__ th2p,
    float* __restrict__ spk_out, float* __restrict__ mem_out,
    int* __restrict__ bar) {
  extern __shared__ __align__(16) char lds_raw[];
  char* ldsA = lds_raw;               // phase-A weight fragments (96KB)
  char* ldsB = lds_raw + LDSA_BYTES;  // phase-B weight fragments (80KB)

  const int lane = threadIdx.x & 31;
  const int wv   = threadIdx.x >> 5;                // 0..15
  const int hl   = lane >> 4;                       // wave half (0/1)
  const int nn   = lane & 15;

  int* bar_cnt = bar;
  int* bar_gen = bar + 1;

  const float th1 = th1p[0];
  const float th2 = th2p[0];

  // Phase A: block -> h-tile (0..31), wave -> b-tile (0..15). 512 pairs.
  const int ht1 = blockIdx.x;
  const int bt1 = wv;
  // Phase B: blocks 0..7 -> o-tile, wave -> b-tile. 128 pairs.
  const bool doB = (blockIdx.x < 8);
  const int ot2 = blockIdx.x;
  const int bt2 = wv;

  // ---- One-time weight staging into LDS, pre-swizzled to fragment order ----
  // Fragment slot: ((ks*4 + g)*32 + lane) * 32 bytes (one v16bf per lane).
  for (int ks = wv; ks < KS1; ks += NTH / 32) {
    const int k0 = ks * 32;
#pragma unroll
    for (int g = 0; g < 4; ++g) {
      const int j = g * Hn + ht1 * 16 + nn;
      v16bf f = (k0 < Dn) ? frag_from_bf16(Wih1b + (size_t)j * Dn, k0, hl)
                          : frag_from_bf16(Whh1b + (size_t)j * Hn, k0 - Dn, hl);
      *(v16bf*)(ldsA + ((size_t)(ks * 4 + g) * 32 + lane) * 32) = f;
    }
  }
  if (doB) {
    for (int ks = wv; ks < KS2; ks += NTH / 32) {
      const int k0 = ks * 32;
#pragma unroll
      for (int g = 0; g < 4; ++g) {
        const int j = g * On + ot2 * 16 + nn;
        v16bf f = (k0 < Hn) ? frag_from_bf16(Wih2b + (size_t)j * Hn, k0, hl)
                            : frag_from_bf16(Whh2b + (size_t)j * On, k0 - Hn, hl);
        *(v16bf*)(ldsB + ((size_t)(ks * 4 + g) * 32 + lane) * 32) = f;
      }
    }
  }
  __syncthreads();

  for (int t = 0; t < Tn; ++t) {
    const __bf16* m1r = (t & 1) ? mem1b1 : mem1b0;
    __bf16*       m1w = (t & 1) ? mem1b0 : mem1b1;
    const __bf16* m2r = (t & 1) ? mem2b1 : mem2b0;
    __bf16*       m2w = (t & 1) ? mem2b0 : mem2b1;

    // ------------------ Phase A: layer-1 gates + SLSTM update ------------------
    {
      v8f acc[4] = {};                   // i, f, g, o gate tiles (16x16 f32 each)
      const int arow = bt1 * 16 + nn;    // A-matrix row owned by this lane
      const float*  xrow = x + ((size_t)arow * Tn + t) * Dn;
      const __bf16* mrow = m1r + (size_t)arow * Hn;

      // Prefetch next timestep's x row (x streams 128MB from HBM)
      if (t + 1 < Tn) __builtin_prefetch(xrow + Dn, 0, 1);

      auto ldsFragA = [&](int ks, int g) -> v16bf {
        return *(const v16bf*)(ldsA + ((size_t)(ks * 4 + g) * 32 + lane) * 32);
      };
      auto loadA1 = [&](int ks) -> v16bf {
        return (ks < Dn / 32) ? frag_from_f32(xrow, ks * 32, hl)
                              : frag_from_bf16(mrow, ks * 32 - Dn, hl);
      };

      // Software-pipelined: next-k A (global) and B (LDS) fragments are
      // issued before the current-k WMMAs consume their own buffers.
      v16bf aN = loadA1(0);
      v16bf bN0 = ldsFragA(0, 0), bN1 = ldsFragA(0, 1);
      v16bf bN2 = ldsFragA(0, 2), bN3 = ldsFragA(0, 3);
#pragma unroll
      for (int ks = 0; ks < KS1; ++ks) {
        v16bf aC = aN, b0 = bN0, b1 = bN1, b2 = bN2, b3 = bN3;
        if (ks + 1 < KS1) {
          aN  = loadA1(ks + 1);
          bN0 = ldsFragA(ks + 1, 0);
          bN1 = ldsFragA(ks + 1, 1);
          bN2 = ldsFragA(ks + 1, 2);
          bN3 = ldsFragA(ks + 1, 3);
        }
        acc[0] = __builtin_amdgcn_wmma_f32_16x16x32_bf16(false, aC, false, b0, (short)0, acc[0], false, false);
        acc[1] = __builtin_amdgcn_wmma_f32_16x16x32_bf16(false, aC, false, b1, (short)0, acc[1], false, false);
        acc[2] = __builtin_amdgcn_wmma_f32_16x16x32_bf16(false, aC, false, b2, (short)0, acc[2], false, false);
        acc[3] = __builtin_amdgcn_wmma_f32_16x16x32_bf16(false, aC, false, b3, (short)0, acc[3], false, false);
      }

      // Pointwise SLSTM update (C layout: vgpr r -> row r + hl*8, lane n -> col)
      const int h = ht1 * 16 + nn;
      const float bi = bias1[h];
      const float bf = bias1[Hn + h];
      const float bg = bias1[2 * Hn + h];
      const float bo = bias1[3 * Hn + h];
#pragma unroll
      for (int r = 0; r < 8; ++r) {
        const int b = bt1 * 16 + r + hl * 8;
        const size_t idx = (size_t)b * Hn + h;
        const float iv = acc[0][r] + bi;
        const float fv = acc[1][r] + bf;
        const float gv = acc[2][r] + bg;
        const float ov = acc[3][r] + bo;
        const float synv = syn1[idx];
        const float memp = mem1f[idx];
        const float cnew = sigm(fv) * synv + sigm(iv) * tanhf(gv);
        const float hnew = sigm(ov) * tanhf(cnew);
        const float mnew = hnew - ((memp > th1) ? th1 : 0.0f);   // STE reset (detached)
        const float spk  = (mnew > th1) ? 1.0f : 0.0f;
        syn1[idx]  = cnew;
        mem1f[idx] = mnew;
        m1w[idx]   = (__bf16)mnew;
        spk1b[idx] = (__bf16)spk;
      }
    }
    grid_barrier(bar_cnt, bar_gen, 2 * t + 1);

    // ------------------ Phase B: layer-2 gates + SLSTM update ------------------
    if (doB) {
      v8f acc[4] = {};
      const int arow = bt2 * 16 + nn;
      const __bf16* srow = spk1b + (size_t)arow * Hn;
      const __bf16* mrow = m2r + (size_t)arow * On;

      auto ldsFragB = [&](int ks, int g) -> v16bf {
        return *(const v16bf*)(ldsB + ((size_t)(ks * 4 + g) * 32 + lane) * 32);
      };
      auto loadA2 = [&](int ks) -> v16bf {
        return (ks < Hn / 32) ? frag_from_bf16(srow, ks * 32, hl)
                              : frag_from_bf16(mrow, ks * 32 - Hn, hl);
      };

      v16bf aN = loadA2(0);
      v16bf bN0 = ldsFragB(0, 0), bN1 = ldsFragB(0, 1);
      v16bf bN2 = ldsFragB(0, 2), bN3 = ldsFragB(0, 3);
#pragma unroll
      for (int ks = 0; ks < KS2; ++ks) {
        v16bf aC = aN, b0 = bN0, b1 = bN1, b2 = bN2, b3 = bN3;
        if (ks + 1 < KS2) {
          aN  = loadA2(ks + 1);
          bN0 = ldsFragB(ks + 1, 0);
          bN1 = ldsFragB(ks + 1, 1);
          bN2 = ldsFragB(ks + 1, 2);
          bN3 = ldsFragB(ks + 1, 3);
        }
        acc[0] = __builtin_amdgcn_wmma_f32_16x16x32_bf16(false, aC, false, b0, (short)0, acc[0], false, false);
        acc[1] = __builtin_amdgcn_wmma_f32_16x16x32_bf16(false, aC, false, b1, (short)0, acc[1], false, false);
        acc[2] = __builtin_amdgcn_wmma_f32_16x16x32_bf16(false, aC, false, b2, (short)0, acc[2], false, false);
        acc[3] = __builtin_amdgcn_wmma_f32_16x16x32_bf16(false, aC, false, b3, (short)0, acc[3], false, false);
      }

      const int o = ot2 * 16 + nn;
      const float bi = bias2[o];
      const float bf = bias2[On + o];
      const float bg = bias2[2 * On + o];
      const float bo = bias2[3 * On + o];
#pragma unroll
      for (int r = 0; r < 8; ++r) {
        const int b = bt2 * 16 + r + hl * 8;
        const size_t idx = (size_t)b * On + o;
        const float iv = acc[0][r] + bi;
        const float fv = acc[1][r] + bf;
        const float gv = acc[2][r] + bg;
        const float ov = acc[3][r] + bo;
        const float synv = syn2[idx];
        const float memp = mem2f[idx];
        const float cnew = sigm(fv) * synv + sigm(iv) * tanhf(gv);
        const float hnew = sigm(ov) * tanhf(cnew);
        const float mnew = hnew - ((memp > th2) ? th2 : 0.0f);
        const float spk  = (mnew > th2) ? 1.0f : 0.0f;
        syn2[idx]  = cnew;
        mem2f[idx] = mnew;
        m2w[idx]   = (__bf16)mnew;
        const size_t oidx = ((size_t)b * Tn + t) * On + o;
        spk_out[oidx] = spk;       // spk_rec [B,T,O]
        mem_out[oidx] = mnew;      // mem_rec [B,T,O]
      }
    }
    grid_barrier(bar_cnt, bar_gen, 2 * t + 2);
  }
}

extern "C" void kernel_launch(void* const* d_in, const int* in_sizes, int n_in,
                              void* d_out, int out_size, void* d_ws, size_t ws_size,
                              hipStream_t stream) {
  (void)in_sizes; (void)n_in; (void)out_size; (void)ws_size;
  const float* x    = (const float*)d_in[0];
  const float* Wih1 = (const float*)d_in[1];
  const float* Whh1 = (const float*)d_in[2];
  const float* bih1 = (const float*)d_in[3];
  const float* bhh1 = (const float*)d_in[4];
  const float* Wih2 = (const float*)d_in[5];
  const float* Whh2 = (const float*)d_in[6];
  const float* bih2 = (const float*)d_in[7];
  const float* bhh2 = (const float*)d_in[8];
  const float* th1  = (const float*)d_in[9];
  const float* th2  = (const float*)d_in[10];

  // Carve workspace (~6.1 MB total, 256B aligned segments)
  char* w = (char*)d_ws;
  auto carve = [&](size_t bytes) -> char* {
    char* p = w;
    w += (bytes + 255) & ~(size_t)255;
    return p;
  };
  __bf16* Wih1b  = (__bf16*)carve((size_t)4 * Hn * Dn * 2);
  __bf16* Whh1b  = (__bf16*)carve((size_t)4 * Hn * Hn * 2);
  __bf16* Wih2b  = (__bf16*)carve((size_t)4 * On * Hn * 2);
  __bf16* Whh2b  = (__bf16*)carve((size_t)4 * On * On * 2);
  float*  bias1  = (float*)carve((size_t)4 * Hn * 4);
  float*  bias2  = (float*)carve((size_t)4 * On * 4);
  float*  syn1   = (float*)carve((size_t)Bn * Hn * 4);
  float*  mem1f  = (float*)carve((size_t)Bn * Hn * 4);
  __bf16* mem1b0 = (__bf16*)carve((size_t)Bn * Hn * 2);
  __bf16* mem1b1 = (__bf16*)carve((size_t)Bn * Hn * 2);
  __bf16* spk1b  = (__bf16*)carve((size_t)Bn * Hn * 2);
  float*  syn2   = (float*)carve((size_t)Bn * On * 4);
  float*  mem2f  = (float*)carve((size_t)Bn * On * 4);
  __bf16* mem2b0 = (__bf16*)carve((size_t)Bn * On * 2);
  __bf16* mem2b1 = (__bf16*)carve((size_t)Bn * On * 2);
  int*    bar    = (int*)carve(256);

  float* spk_out = (float*)d_out;
  float* mem_out = spk_out + (size_t)Bn * Tn * On;

  slstm_init<<<dim3(1024), dim3(256), 0, stream>>>(
      Wih1, Whh1, bih1, bhh1, Wih2, Whh2, bih2, bhh2,
      Wih1b, Whh1b, Wih2b, Whh2b, bias1, bias2,
      syn1, mem1f, mem1b0, syn2, mem2f, mem2b0, bar);

  slstm_persistent<<<dim3(NWG), dim3(NTH), LDS_BYTES, stream>>>(
      x, Wih1b, Whh1b, Wih2b, Whh2b, bias1, bias2,
      syn1, mem1f, mem1b0, mem1b1, spk1b,
      syn2, mem2f, mem2b0, mem2b1,
      th1, th2, spk_out, mem_out, bar);
}
